// GraphHop_70196945486069
// MI455X (gfx1250) — matrix operand order repaired and checked by reading, not compiled, not run on hardware
//
#include <hip/hip_runtime.h>

#define DEVFN __device__ __forceinline__

typedef unsigned short bfraw;
typedef __attribute__((ext_vector_type(16))) __bf16 v16bf;
typedef __attribute__((ext_vector_type(8))) float v8f;
typedef __attribute__((ext_vector_type(4))) unsigned int u32x4;

union Frag16 { v16bf v; u32x4 q[2]; bfraw e[16]; };

constexpr int NN  = 100000;
constexpr int HH  = 768;
constexpr int EE  = 400000;
constexpr int NGP = 64;
constexpr int EMB = 512;
constexpr int E2  = EE + NN;   // edges + self loops

DEVFN bfraw f2b(float f) {
  unsigned u = __float_as_uint(f);
  u += 0x7FFFu + ((u >> 16) & 1u);          // round-to-nearest-even
  return (bfraw)(u >> 16);
}
DEVFN float b2f(bfraw b) { return __uint_as_float(((unsigned)b) << 16); }
// order-preserving f32 -> u32 map for atomicMax-based segment-max
DEVFN unsigned mapf(float f) {
  unsigned u = __float_as_uint(f);
  return (u & 0x80000000u) ? ~u : (u | 0x80000000u);
}
DEVFN float unmapf(unsigned m) {
  unsigned u = (m & 0x80000000u) ? (m & 0x7FFFFFFFu) : ~m;
  return __uint_as_float(u);
}

// ---------------------------------------------------------------------------
// WMMA bf16 GEMM:  C[M,Nc] = act( A[M,K](bf16) @ Wt[Nc,K]^T (bf16) + bias )
// A may be split in K (A0 rows of length K0, A1 rows of length K-K0) to fuse
// the concat([h,g]) input of the GAT projection without materializing it.
// Non-split callers pass A1 == A0 and K0 == K.  K must be a multiple of 64.
// Block: 256 threads = 8 waves (4 M-waves x 2 N-waves). Block tile 64x128.
// Wave tile 16x64 -> 4 f32 accumulators. K step 32 per v_wmma_f32_16x16x32_bf16.
// Ping-pong double buffering with NAMED fragment sets (constant indexing only,
// so everything stays in VGPRs). The in-loop loads are UNCONDITIONAL and the
// final K-step is peeled, giving a single control path through the loop body
// so the compiler can emit partial s_wait_loadcnt for both MMA groups.
// Fragment layouts follow CDNA5 ISA 7.12.2:
//   A 16x32 bf16: lane<16 holds row m, K {0..7,16..23}; lane>=16 K {8..15,24..31}
//     -> two contiguous 16B loads at +8*half and +16+8*half.
//   B 32x16 bf16 (from pre-transposed Wt[n][k]): lane holds col n=lane&15,
//     K {8*half..+7, 16+8*half..+7} (same as A) -> two 16B loads.
//   C/D 16x16 f32: VGPR i -> row i+8*half, col lane&15.
// ---------------------------------------------------------------------------
__global__ void __launch_bounds__(256) k_gemm_bf16(
    const bfraw* __restrict__ A0, const bfraw* __restrict__ A1, int K0,
    const bfraw* __restrict__ Wt, const float* __restrict__ bias,
    int M, int K, int Nc,
    float* __restrict__ Cf, bfraw* __restrict__ Cb, int relu)
{
  const int lane = (int)(threadIdx.x & 31u);
  const int wave = (int)(threadIdx.x >> 5u);
  const int mw = wave >> 1, nw = wave & 1;
  const int half = lane >> 4, l16 = lane & 15;
  const int rowBase = (int)blockIdx.x * 64 + mw * 16;
  const int colBase = (int)blockIdx.y * 128 + nw * 64;
  int aRow = rowBase + l16;
  if (aRow >= M) aRow = M - 1;                 // clamp (dup loads, stores masked)

  // pre-biased per-lane A row pointers: index both with plain kk
  const bfraw* pA0 = A0 + (size_t)aRow * (size_t)K0 + half * 8;
  const bfraw* pA1 = A1 + (size_t)aRow * (size_t)(K - K0) + half * 8 - K0;

  // per-tile B base pointers (advance by kk elements each step)
  const bfraw* bp0 = Wt + (size_t)(colBase + 0  + l16) * (size_t)K + half * 8;
  const bfraw* bp1 = Wt + (size_t)(colBase + 16 + l16) * (size_t)K + half * 8;
  const bfraw* bp2 = Wt + (size_t)(colBase + 32 + l16) * (size_t)K + half * 8;
  const bfraw* bp3 = Wt + (size_t)(colBase + 48 + l16) * (size_t)K + half * 8;

  v8f acc0, acc1, acc2, acc3;
#pragma unroll
  for (int i = 0; i < 8; ++i) { acc0[i] = 0.f; acc1[i] = 0.f; acc2[i] = 0.f; acc3[i] = 0.f; }

  Frag16 fa0, fa1;
  Frag16 fb00, fb01, fb02, fb03;   // buffer 0, tiles 0..3
  Frag16 fb10, fb11, fb12, fb13;   // buffer 1, tiles 0..3

#define LOAD_A(kk, F)                                                        \
  { const bfraw* ar = ((kk) < K0 ? pA0 : pA1) + (kk);                        \
    (F).q[0] = *(const u32x4*)(const void*)(ar);                             \
    (F).q[1] = *(const u32x4*)(const void*)(ar + 16);                        \
    __builtin_prefetch((const void*)(ar + 64), 0, 1); }
#define LOAD_B(kk, F0, F1, F2, F3)                                           \
  { (F0).q[0] = *(const u32x4*)(const void*)(bp0 + (kk));                    \
    (F0).q[1] = *(const u32x4*)(const void*)(bp0 + (kk) + 16);               \
    (F1).q[0] = *(const u32x4*)(const void*)(bp1 + (kk));                    \
    (F1).q[1] = *(const u32x4*)(const void*)(bp1 + (kk) + 16);               \
    (F2).q[0] = *(const u32x4*)(const void*)(bp2 + (kk));                    \
    (F2).q[1] = *(const u32x4*)(const void*)(bp2 + (kk) + 16);               \
    (F3).q[0] = *(const u32x4*)(const void*)(bp3 + (kk));                    \
    (F3).q[1] = *(const u32x4*)(const void*)(bp3 + (kk) + 16); }
#define MMA(FA, F0, F1, F2, F3)                                              \
  { acc0 = __builtin_amdgcn_wmma_f32_16x16x32_bf16(false, (FA).v, false, (F0).v, (short)0, acc0, false, false); \
    acc1 = __builtin_amdgcn_wmma_f32_16x16x32_bf16(false, (FA).v, false, (F1).v, (short)0, acc1, false, false); \
    acc2 = __builtin_amdgcn_wmma_f32_16x16x32_bf16(false, (FA).v, false, (F2).v, (short)0, acc2, false, false); \
    acc3 = __builtin_amdgcn_wmma_f32_16x16x32_bf16(false, (FA).v, false, (F3).v, (short)0, acc3, false, false); }

  LOAD_A(0, fa0); LOAD_B(0, fb00, fb01, fb02, fb03);
  LOAD_A(32, fa1); LOAD_B(32, fb10, fb11, fb12, fb13);

  // steady state: single control path, loads unconditional
  int kk = 0;
  for (; kk + 64 < K; kk += 64) {            // K % 64 == 0
    MMA(fa0, fb00, fb01, fb02, fb03);
    LOAD_A(kk + 64, fa0); LOAD_B(kk + 64, fb00, fb01, fb02, fb03);
    MMA(fa1, fb10, fb11, fb12, fb13);
    LOAD_A(kk + 96, fa1); LOAD_B(kk + 96, fb10, fb11, fb12, fb13);
  }
  // peeled final K-step pair: no further loads
  MMA(fa0, fb00, fb01, fb02, fb03);
  MMA(fa1, fb10, fb11, fb12, fb13);
#undef LOAD_A
#undef LOAD_B
#undef MMA

#pragma unroll
  for (int t = 0; t < 4; ++t) {
    v8f acc = (t == 0) ? acc0 : (t == 1) ? acc1 : (t == 2) ? acc2 : acc3;
    const int n = colBase + t * 16 + l16;
    const float bv = bias ? bias[n] : 0.0f;
#pragma unroll
    for (int i = 0; i < 8; ++i) {
      const int r = rowBase + i + 8 * half;
      if (r < M) {
        float v = acc[i] + bv;
        if (relu) v = fmaxf(v, 0.0f);
        const size_t o = (size_t)r * (size_t)Nc + (size_t)n;
        if (Cf) Cf[o] = v;
        if (Cb) Cb[o] = f2b(v);
      }
    }
  }
}

// --------------------------- auxiliary kernels -----------------------------

// weight transpose+cast: Wt[n*K + k] = bf16( W[k*ld + col0 + n] )
__global__ void k_transpose_cast(const float* __restrict__ W, int ld, int col0,
                                 int K, int Nc, bfraw* __restrict__ Wt)
{
  long idx = (long)blockIdx.x * blockDim.x + threadIdx.x;
  if (idx >= (long)K * Nc) return;
  int n = (int)(idx / K), k = (int)(idx % K);
  Wt[(size_t)n * K + k] = f2b(W[(size_t)k * ld + col0 + n]);
}

// projection: h = x @ proj_W + b  (K=3, pure VALU)
__global__ void k_proj(const float* __restrict__ x, const float* __restrict__ W,
                       const float* __restrict__ b, float* __restrict__ hf,
                       bfraw* __restrict__ hb)
{
  long idx = (long)blockIdx.x * blockDim.x + threadIdx.x;
  if (idx >= (long)NN * HH) return;
  int n = (int)(idx / HH), c = (int)(idx % HH);
  float v = b[c] + x[n * 3 + 0] * W[c] + x[n * 3 + 1] * W[HH + c] +
            x[n * 3 + 2] * W[2 * HH + c];
  hf[idx] = v;
  hb[idx] = f2b(v);
}

// GIN aggregation: agg[dst] += h[src]   (thread per (edge, 4-float chunk))
__global__ void k_scatter_gin(const int* __restrict__ src,
                              const int* __restrict__ dst,
                              const float* __restrict__ hf,
                              float* __restrict__ agg)
{
  long idx = (long)blockIdx.x * blockDim.x + threadIdx.x;
  if (idx >= (long)EE * (HH / 4)) return;
  int e = (int)(idx / (HH / 4)), c4 = (int)(idx % (HH / 4));
  int s = src[e], d = dst[e];
  const float4 v = *(const float4*)(hf + (size_t)s * HH + c4 * 4);
  float* p = agg + (size_t)d * HH + c4 * 4;
  atomicAdd(p + 0, v.x); atomicAdd(p + 1, v.y);
  atomicAdd(p + 2, v.z); atomicAdd(p + 3, v.w);
}

// u = bf16(h + agg)
__global__ void k_add_u(const float* __restrict__ hf, const float* __restrict__ agg,
                        bfraw* __restrict__ u)
{
  long idx = (long)blockIdx.x * blockDim.x + threadIdx.x;
  if (idx >= (long)NN * HH) return;
  u[idx] = f2b(hf[idx] + agg[idx]);
}

// per-node per-head attention logits a_s, a_d
__global__ void k_att(const bfraw* __restrict__ xw, const float* __restrict__ asrc,
                      const float* __restrict__ adst, float* __restrict__ a_s,
                      float* __restrict__ a_d)
{
  long idx = (long)blockIdx.x * blockDim.x + threadIdx.x;
  if (idx >= (long)NN * 4) return;
  int n = (int)(idx >> 2), hd = (int)(idx & 3);
  const bfraw* xr = xw + (size_t)n * HH + hd * 192;
  float s = 0.f, d = 0.f;
  for (int j = 0; j < 192; ++j) {
    float f = b2f(xr[j]);
    s += f * asrc[hd * 192 + j];
    d += f * adst[hd * 192 + j];
  }
  a_s[idx] = s; a_d[idx] = d;
}

DEVFN void edge_sd(int e, const int* src, const int* dst, int& s, int& d) {
  if (e < EE) { s = src[e]; d = dst[e]; } else { s = e - EE; d = e - EE; }
}
DEVFN float leaky(float x) { return x > 0.f ? x : 0.2f * x; }

__global__ void k_emax(const int* __restrict__ src, const int* __restrict__ dst,
                       const float* __restrict__ a_s, const float* __restrict__ a_d,
                       unsigned* __restrict__ emax)
{
  long idx = (long)blockIdx.x * blockDim.x + threadIdx.x;
  if (idx >= (long)E2 * 4) return;
  int e = (int)(idx >> 2), h = (int)(idx & 3), s, d;
  edge_sd(e, src, dst, s, d);
  float ev = leaky(a_s[s * 4 + h] + a_d[d * 4 + h]);
  atomicMax(emax + d * 4 + h, mapf(ev));
}

__global__ void k_den(const int* __restrict__ src, const int* __restrict__ dst,
                      const float* __restrict__ a_s, const float* __restrict__ a_d,
                      const unsigned* __restrict__ emax, float* __restrict__ den)
{
  long idx = (long)blockIdx.x * blockDim.x + threadIdx.x;
  if (idx >= (long)E2 * 4) return;
  int e = (int)(idx >> 2), h = (int)(idx & 3), s, d;
  edge_sd(e, src, dst, s, d);
  float ev = leaky(a_s[s * 4 + h] + a_d[d * 4 + h]);
  atomicAdd(den + d * 4 + h, __expf(ev - unmapf(emax[d * 4 + h])));
}

// gat[d, h*192 + c*16 .. +16] += alpha(e,h) * xw[s, ...]
__global__ void k_gat_scatter(const int* __restrict__ src, const int* __restrict__ dst,
                              const float* __restrict__ a_s, const float* __restrict__ a_d,
                              const unsigned* __restrict__ emax, const float* __restrict__ den,
                              const bfraw* __restrict__ xw, float* __restrict__ gat)
{
  long idx = (long)blockIdx.x * blockDim.x + threadIdx.x;
  if (idx >= (long)E2 * 48) return;
  int e = (int)(idx / 48), rem = (int)(idx % 48);
  int h = rem / 12;                            // head 0..3
  int c = rem % 12;                            // 16-feature chunk 0..11
  int s, d;
  edge_sd(e, src, dst, s, d);
  float ev = leaky(a_s[s * 4 + h] + a_d[d * 4 + h]);
  float ex = __expf(ev - unmapf(emax[d * 4 + h]));
  float alpha = ex / (den[d * 4 + h] + 1e-16f);
  Frag16 buf;
  const bfraw* xp = xw + (size_t)s * HH + h * 192 + c * 16;
  buf.q[0] = *(const u32x4*)(const void*)(xp);
  buf.q[1] = *(const u32x4*)(const void*)(xp + 8);
  float* gp = gat + (size_t)d * HH + h * 192 + c * 16;
#pragma unroll
  for (int j = 0; j < 16; ++j) atomicAdd(gp + j, b2f(buf.e[j]) * alpha);
}

// gat_bf16 = bf16(relu(gat_f + gat_b))
__global__ void k_gat_final(const float* __restrict__ gf, const float* __restrict__ gb,
                            bfraw* __restrict__ out)
{
  long idx = (long)blockIdx.x * blockDim.x + threadIdx.x;
  if (idx >= (long)NN * HH) return;
  int c = (int)(idx % HH);
  out[idx] = f2b(fmaxf(gf[idx] + gb[c], 0.0f));
}

// per-node per-head 2-key attention: o = softmax(q.k/sqrt(96)) . v
__global__ void k_mha(const bfraw* __restrict__ q, const bfraw* __restrict__ k0,
                      const bfraw* __restrict__ k1, const bfraw* __restrict__ v0,
                      const bfraw* __restrict__ v1, bfraw* __restrict__ o)
{
  long idx = (long)blockIdx.x * blockDim.x + threadIdx.x;
  if (idx >= (long)NN * 8) return;
  int n = (int)(idx >> 3), h = (int)(idx & 7);
  size_t base = (size_t)n * HH + h * 96;
  float s0 = 0.f, s1 = 0.f;
  for (int j = 0; j < 96; ++j) {
    float fq = b2f(q[base + j]);
    s0 += fq * b2f(k0[base + j]);
    s1 += fq * b2f(k1[base + j]);
  }
  const float inv = 0.102062072615966f;  // 1/sqrt(96)
  s0 *= inv; s1 *= inv;
  float m = fmaxf(s0, s1);
  float e0 = __expf(s0 - m), e1 = __expf(s1 - m);
  float a0 = e0 / (e0 + e1), a1 = 1.0f - a0;
  for (int j = 0; j < 96; ++j)
    o[base + j] = f2b(a0 * b2f(v0[base + j]) + a1 * b2f(v1[base + j]));
}

__global__ void k_count(const int* __restrict__ batch, float* __restrict__ cnt)
{
  long n = (long)blockIdx.x * blockDim.x + threadIdx.x;
  if (n >= NN) return;
  atomicAdd(cnt + batch[n], 1.0f);
}

__global__ void k_pool_sum(const bfraw* __restrict__ o2, const int* __restrict__ batch,
                           float* __restrict__ sums)
{
  long idx = (long)blockIdx.x * blockDim.x + threadIdx.x;
  if (idx >= (long)NN * (HH / 4)) return;
  int n = (int)(idx / (HH / 4)), c4 = (int)(idx % (HH / 4));
  int b = batch[n];
  const bfraw* p = o2 + (size_t)n * HH + c4 * 4;
  float* sp = sums + (size_t)b * HH + c4 * 4;
#pragma unroll
  for (int j = 0; j < 4; ++j) atomicAdd(sp + j, b2f(p[j]));
}

__global__ void k_pool_div(const float* __restrict__ sums, const float* __restrict__ cnt,
                           bfraw* __restrict__ pooled)
{
  long idx = (long)blockIdx.x * blockDim.x + threadIdx.x;
  if (idx >= (long)NGP * HH) return;
  int g = (int)(idx / HH);
  pooled[idx] = f2b(sums[idx] / fmaxf(cnt[g], 1.0f));
}

// ------------------------------- launcher ----------------------------------

extern "C" void kernel_launch(void* const* d_in, const int* in_sizes, int n_in,
                              void* d_out, int out_size, void* d_ws, size_t ws_size,
                              hipStream_t stream)
{
  (void)in_sizes; (void)n_in; (void)out_size; (void)ws_size;
  const float* x       = (const float*)d_in[0];
  const int*   eidx    = (const int*)d_in[1];
  const int*   batch   = (const int*)d_in[2];
  const float* proj_W  = (const float*)d_in[3];
  const float* proj_b  = (const float*)d_in[4];
  const float* gin1_W  = (const float*)d_in[5];
  const float* gin1_b  = (const float*)d_in[6];
  const float* gin2_W  = (const float*)d_in[7];
  const float* gin2_b  = (const float*)d_in[8];
  const float* gat_W   = (const float*)d_in[9];
  const float* att_src = (const float*)d_in[10];
  const float* att_dst = (const float*)d_in[11];
  const float* gat_b   = (const float*)d_in[12];
  const float* in_W    = (const float*)d_in[13];
  const float* in_b    = (const float*)d_in[14];
  const float* out_W   = (const float*)d_in[15];
  const float* out_b   = (const float*)d_in[16];
  const float* fc_W    = (const float*)d_in[17];
  const float* fc_b    = (const float*)d_in[18];
  float* out = (float*)d_out;

  const int* src = eidx;            // edge_index[0]
  const int* dst = eidx + EE;       // edge_index[1]

  const size_t SZ_F = (size_t)NN * HH * 4;   // 307.2 MB
  const size_t SZ_B = (size_t)NN * HH * 2;   // 153.6 MB
  char* W0 = (char*)d_ws;
  // big regions with explicit reuse over the pipeline
  const size_t oA = 0;               // h_f32 -> gat_f32 -> v0|v1 (SZ_F)
  const size_t oB = oA + SZ_F;       // h_bf16                     (SZ_B)
  const size_t oC = oB + SZ_B;       // agg -> [gat_bf16 | o_bf16] (SZ_F)
  const size_t oD = oC + SZ_F;       // u -> q                     (SZ_B)
  const size_t oE = oD + SZ_B;       // g1 -> k0 -> o2             (SZ_B)
  const size_t oF = oE + SZ_B;       // g                          (SZ_B)
  const size_t oG = oF + SZ_B;       // xw -> k1                   (SZ_B)
  size_t cur = oG + SZ_B;            // small pool
  auto al = [&](size_t bytes) { size_t o = cur; cur = (cur + bytes + 255) & ~(size_t)255; return o; };
  const size_t o_as    = al((size_t)NN * 4 * 4);
  const size_t o_ad    = al((size_t)NN * 4 * 4);
  const size_t o_emax  = al((size_t)NN * 4 * 4);
  const size_t o_den   = al((size_t)NN * 4 * 4);
  const size_t o_sums  = al((size_t)NGP * HH * 4);
  const size_t o_cnt   = al((size_t)NGP * 4);
  const size_t o_pool  = al((size_t)NGP * HH * 2);
  const size_t o_wg1   = al((size_t)HH * HH * 2);
  const size_t o_wg2   = al((size_t)HH * HH * 2);
  const size_t o_wgat  = al((size_t)HH * 2 * HH * 2);
  const size_t o_wq    = al((size_t)HH * HH * 2);
  const size_t o_wk    = al((size_t)HH * HH * 2);
  const size_t o_wv    = al((size_t)HH * HH * 2);
  const size_t o_wo    = al((size_t)HH * HH * 2);
  const size_t o_wfc   = al((size_t)HH * EMB * 2);

  float*  h_f   = (float*)(W0 + oA);
  bfraw*  h_b   = (bfraw*)(W0 + oB);
  float*  agg   = (float*)(W0 + oC);
  bfraw*  u_b   = (bfraw*)(W0 + oD);
  bfraw*  g1_b  = (bfraw*)(W0 + oE);
  bfraw*  g_b   = (bfraw*)(W0 + oF);
  bfraw*  xw_b  = (bfraw*)(W0 + oG);
  float*  gat_f = (float*)(W0 + oA);
  bfraw*  gat_bb= (bfraw*)(W0 + oC);
  bfraw*  o_bb  = (bfraw*)(W0 + oC + SZ_B);
  bfraw*  q_b   = (bfraw*)(W0 + oD);
  bfraw*  k0_b  = (bfraw*)(W0 + oE);
  bfraw*  k1_b  = (bfraw*)(W0 + oG);
  bfraw*  v0_b  = (bfraw*)(W0 + oA);
  bfraw*  v1_b  = (bfraw*)(W0 + oA + SZ_B);
  bfraw*  o2_b  = (bfraw*)(W0 + oE);
  float*  a_s   = (float*)(W0 + o_as);
  float*  a_d   = (float*)(W0 + o_ad);
  unsigned* emax= (unsigned*)(W0 + o_emax);
  float*  den   = (float*)(W0 + o_den);
  float*  sums  = (float*)(W0 + o_sums);
  float*  cntb  = (float*)(W0 + o_cnt);
  bfraw*  pooled= (bfraw*)(W0 + o_pool);
  bfraw*  wg1   = (bfraw*)(W0 + o_wg1);
  bfraw*  wg2   = (bfraw*)(W0 + o_wg2);
  bfraw*  wgat  = (bfraw*)(W0 + o_wgat);
  bfraw*  wq    = (bfraw*)(W0 + o_wq);
  bfraw*  wk    = (bfraw*)(W0 + o_wk);
  bfraw*  wv    = (bfraw*)(W0 + o_wv);
  bfraw*  wo    = (bfraw*)(W0 + o_wo);
  bfraw*  wfc   = (bfraw*)(W0 + o_wfc);

  auto NB = [](long n) { return (unsigned)((n + 255) / 256); };
  const dim3 B256(256);

  // weight transpose+cast (tiny; weights then live in L2)
  k_transpose_cast<<<NB((long)HH * HH), B256, 0, stream>>>(gin1_W, HH, 0, HH, HH, wg1);
  k_transpose_cast<<<NB((long)HH * HH), B256, 0, stream>>>(gin2_W, HH, 0, HH, HH, wg2);
  k_transpose_cast<<<NB((long)2 * HH * HH), B256, 0, stream>>>(gat_W, HH, 0, 2 * HH, HH, wgat);
  k_transpose_cast<<<NB((long)HH * HH), B256, 0, stream>>>(in_W, 3 * HH, 0,      HH, HH, wq);
  k_transpose_cast<<<NB((long)HH * HH), B256, 0, stream>>>(in_W, 3 * HH, HH,     HH, HH, wk);
  k_transpose_cast<<<NB((long)HH * HH), B256, 0, stream>>>(in_W, 3 * HH, 2 * HH, HH, HH, wv);
  k_transpose_cast<<<NB((long)HH * HH), B256, 0, stream>>>(out_W, HH, 0, HH, HH, wo);
  k_transpose_cast<<<NB((long)HH * EMB), B256, 0, stream>>>(fc_W, EMB, 0, HH, EMB, wfc);

  hipMemsetAsync(agg,  0, SZ_F, stream);
  hipMemsetAsync(emax, 0, (size_t)NN * 4 * 4, stream);   // 0 maps to -inf-ish
  hipMemsetAsync(den,  0, (size_t)NN * 4 * 4, stream);
  hipMemsetAsync(sums, 0, (size_t)NGP * HH * 4, stream);
  hipMemsetAsync(cntb, 0, (size_t)NGP * 4, stream);

  // 1) projection
  k_proj<<<NB((long)NN * HH), B256, 0, stream>>>(x, proj_W, proj_b, h_f, h_b);
  // 2) GIN aggregation + MLP
  k_scatter_gin<<<NB((long)EE * (HH / 4)), B256, 0, stream>>>(src, dst, h_f, agg);
  k_add_u<<<NB((long)NN * HH), B256, 0, stream>>>(h_f, agg, u_b);

  const dim3 gN((NN + 63) / 64, HH / 128);
  k_gemm_bf16<<<gN, B256, 0, stream>>>(u_b,  u_b,  HH, wg1, gin1_b, NN, HH, HH, nullptr, g1_b, 1);
  k_gemm_bf16<<<gN, B256, 0, stream>>>(g1_b, g1_b, HH, wg2, gin2_b, NN, HH, HH, nullptr, g_b, 1);
  // 3) GAT projection on concat([h,g]) via split-K A operand
  k_gemm_bf16<<<gN, B256, 0, stream>>>(h_b, g_b, HH, wgat, nullptr, NN, 2 * HH, HH, nullptr, xw_b, 0);
  // 4) edge softmax + aggregation
  k_att<<<NB((long)NN * 4), B256, 0, stream>>>(xw_b, att_src, att_dst, a_s, a_d);
  hipMemsetAsync(gat_f, 0, SZ_F, stream);                // region A is free now
  k_emax<<<NB((long)E2 * 4), B256, 0, stream>>>(src, dst, a_s, a_d, emax);
  k_den<<<NB((long)E2 * 4), B256, 0, stream>>>(src, dst, a_s, a_d, emax, den);
  k_gat_scatter<<<NB((long)E2 * 48), B256, 0, stream>>>(src, dst, a_s, a_d, emax, den, xw_b, gat_f);
  k_gat_final<<<NB((long)NN * HH), B256, 0, stream>>>(gat_f, gat_b, gat_bb);
  // 5) MHA projections (q from gat; k/v from the two hop reps h,g)
  k_gemm_bf16<<<gN, B256, 0, stream>>>(gat_bb, gat_bb, HH, wq, in_b,          NN, HH, HH, nullptr, q_b, 0);
  k_gemm_bf16<<<gN, B256, 0, stream>>>(h_b,    h_b,    HH, wk, in_b + HH,     NN, HH, HH, nullptr, k0_b, 0);
  k_gemm_bf16<<<gN, B256, 0, stream>>>(g_b,    g_b,    HH, wk, in_b + HH,     NN, HH, HH, nullptr, k1_b, 0);
  k_gemm_bf16<<<gN, B256, 0, stream>>>(h_b,    h_b,    HH, wv, in_b + 2 * HH, NN, HH, HH, nullptr, v0_b, 0);
  k_gemm_bf16<<<gN, B256, 0, stream>>>(g_b,    g_b,    HH, wv, in_b + 2 * HH, NN, HH, HH, nullptr, v1_b, 0);
  k_mha<<<NB((long)NN * 8), B256, 0, stream>>>(q_b, k0_b, k1_b, v0_b, v1_b, o_bb);
  k_gemm_bf16<<<gN, B256, 0, stream>>>(o_bb, o_bb, HH, wo, out_b, NN, HH, HH, nullptr, o2_b, 0);
  // 6) mean pool + fc
  k_count<<<NB((long)NN), B256, 0, stream>>>(batch, cntb);
  k_pool_sum<<<NB((long)NN * (HH / 4)), B256, 0, stream>>>(o2_b, batch, sums);
  k_pool_div<<<NB((long)NGP * HH), B256, 0, stream>>>(sums, cntb, pooled);
  const dim3 gFC(1, EMB / 128);
  k_gemm_bf16<<<gFC, B256, 0, stream>>>(pooled, pooled, HH, wfc, fc_b, NGP, HH, EMB, out, nullptr, 0);
}
